// PillarFeatureNet2_MSG_68118181314865
// MI455X (gfx1250) — matrix operand order repaired and checked by reading, not compiled, not run on hardware
//
#include <hip/hip_runtime.h>

typedef _Float16 v16h __attribute__((ext_vector_type(16)));
typedef _Float16 v8h  __attribute__((ext_vector_type(8)));
typedef float    v8f  __attribute__((ext_vector_type(8)));

#define P_TOT 12000
#define N_PTS 100
#define C_IN_ 4
#define NCENT 10
#define SA1_CH 288            // 259 channels padded to 9*32
#define PN ((size_t)P_TOT * N_PTS)

// ---------------------------------------------------------------------------
// WMMA fragment loaders (CDNA5 wave32 layouts, CDNA5 ISA)
// A (16x32 f16): lane<16 -> row M=lane, halves K = kb+{0..7, 16..23}
//                lane>=16 -> row M=lane-16, halves K = kb+{8..15, 24..31}
// B (32x16 f16): lane<16 -> col N=lane, halves K = kb+{0..7, 16..23}
//                lane>=16 -> col N=lane-16, halves K = kb+{8..15, 24..31}
// C/D (16x16 f32): vgpr r -> M = r + (lane>=16 ? 8 : 0), N = lane&15
// ---------------------------------------------------------------------------
__device__ __forceinline__ v16h load_afrag(const _Float16* buf, int stride,
                                           int m0, int kbase, int lane) {
  const _Float16* p = buf + (size_t)(m0 + (lane & 15)) * stride + kbase + ((lane & 16) ? 8 : 0);
  v8h lo = *(const v8h*)p;
  v8h hi = *(const v8h*)(p + 16);
  v16h r;
#pragma unroll
  for (int i = 0; i < 8; ++i) { r[i] = lo[i]; r[i + 8] = hi[i]; }
  return r;
}

__device__ __forceinline__ v16h load_bfrag(const _Float16* w, int kpad,
                                           int n0, int kbase, int lane) {
  const _Float16* p = w + (size_t)(n0 + (lane & 15)) * kpad + kbase + ((lane & 16) ? 8 : 0);
  v8h lo = *(const v8h*)p;
  v8h hi = *(const v8h*)(p + 16);
  v16h r;
#pragma unroll
  for (int i = 0; i < 8; ++i) { r[i] = lo[i]; r[i + 8] = hi[i]; }
  return r;
}

// Block-cooperative GEMM: out[M,N] = relu(in[M,K] * W^T + bias), f16 out.
// A from LDS, B (weights, N x kpad row-major f16) from global (L2/WGP$ cached).
__device__ __forceinline__ void block_gemm_relu(
    const _Float16* aBuf, int aStride,
    const _Float16* wG, int kpad,
    const float* bias,
    _Float16* oBuf, int oStride,
    int Mt, int Nt, int Kc)
{
  const int lane = threadIdx.x & 31;
  const int wave = threadIdx.x >> 5;
  const int nTiles = Mt * Nt;
  for (int t = wave; t < nTiles; t += 8) {
    const int mt = t / Nt;
    const int nt = t - mt * Nt;
    v8f acc = {0.f, 0.f, 0.f, 0.f, 0.f, 0.f, 0.f, 0.f};
    for (int kc = 0; kc < Kc; ++kc) {
      v16h a = load_afrag(aBuf, aStride, mt * 16, kc * 32, lane);
      v16h b = load_bfrag(wG, kpad, nt * 16, kc * 32, lane);
      acc = __builtin_amdgcn_wmma_f32_16x16x32_f16(false, a, false, b,
                                                   (short)0, acc, false, false);
    }
    const int col = nt * 16 + (lane & 15);
    const float bv = bias[col];
    const int rbase = mt * 16 + ((lane & 16) ? 8 : 0);
#pragma unroll
    for (int r = 0; r < 8; ++r) {
      float v = fmaxf(acc[r] + bv, 0.f);
      oBuf[(size_t)(rbase + r) * oStride + col] = (_Float16)v;
    }
  }
}

// async global->LDS B128 (ASYNCcnt path); laddr = LDS byte address, 16B chunks
__device__ __forceinline__ void async_b128(unsigned laddr, const void* gaddr) {
  asm volatile("global_load_async_to_lds_b128 %0, %1, off"
               :: "v"(laddr), "v"(gaddr) : "memory");
}
__device__ __forceinline__ void async_wait0() {
  asm volatile("s_wait_asynccnt 0x0" ::: "memory");
}
__device__ __forceinline__ unsigned lds_addr(const void* p) {
  return (unsigned)(uintptr_t)p;   // generic LDS ptr: offset lives in low 32 bits
}

// ---------------------------------------------------------------------------
// K0: fold BN into weights (f16, K padded to mult of 32) and bias (f32)
// ---------------------------------------------------------------------------
__global__ void k_fold(const float* __restrict__ w, const float* __restrict__ b,
                       const float* __restrict__ g, const float* __restrict__ be,
                       const float* __restrict__ mn, const float* __restrict__ vr,
                       int N, int K, int kpad,
                       _Float16* __restrict__ wOut, float* __restrict__ bOut) {
  int idx = blockIdx.x * blockDim.x + threadIdx.x;
  int total = N * kpad;
  if (idx >= total) return;
  int n = idx / kpad;
  int kk = idx - n * kpad;
  float s = g[n] * rsqrtf(vr[n] + 1e-5f);
  float v = (kk < K) ? w[n * K + kk] * s : 0.f;
  wOut[idx] = (_Float16)v;
  if (kk == 0) bOut[n] = (b[n] - mn[n]) * s + be[n];
}

// zero-fill (v8h chunks)
__global__ void k_zero(_Float16* __restrict__ p, size_t nChunks) {
  size_t i = (size_t)blockIdx.x * blockDim.x + threadIdx.x;
  v8h z = {};
  for (; i < nChunks; i += (size_t)gridDim.x * blockDim.x)
    ((v8h*)p)[i] = z;
}

// ---------------------------------------------------------------------------
// K1: per-pillar feature build: xyz (3 SoA planes f32, masked) + data (6ch f16)
// ---------------------------------------------------------------------------
__global__ __launch_bounds__(128) void k_pre(
    const float* __restrict__ feat, const int* __restrict__ nvox,
    const int* __restrict__ coors,
    float* __restrict__ xyzOut, _Float16* __restrict__ dataOut)
{
  const int p = blockIdx.x;
  const int t = threadIdx.x;
  __shared__ float red[3][128];
  float sx = 0.f, sy = 0.f, sz = 0.f;
  if (t < N_PTS) {
    const float* f = feat + ((size_t)p * N_PTS + t) * C_IN_;
    sx = f[0]; sy = f[1]; sz = f[2];
  }
  red[0][t] = sx; red[1][t] = sy; red[2][t] = sz;
  __syncthreads();
  for (int s = 64; s > 0; s >>= 1) {
    if (t < s) {
      red[0][t] += red[0][t + s];
      red[1][t] += red[1][t + s];
      red[2][t] += red[2][t + s];
    }
    __syncthreads();
  }
  const int nvi = nvox[p];
  const float inv = 1.f / (float)nvi;
  const float mx = red[0][0] * inv, my = red[1][0] * inv, mz = red[2][0] * inv;
  const float ccx = (float)coors[p * 4 + 3] * 0.2f + 0.1f;
  const float ccy = (float)coors[p * 4 + 2] * 0.2f + (0.1f - 40.f);
  if (t < N_PTS) {
    const float* f = feat + ((size_t)p * N_PTS + t) * C_IN_;
    const float m = (t < nvi) ? 1.f : 0.f;
    const float x = f[0], y = f[1], z = f[2], w = f[3];
    size_t base = (size_t)p * N_PTS + t;
    xyzOut[0 * PN + base] = x * m;
    xyzOut[1 * PN + base] = y * m;
    xyzOut[2 * PN + base] = z * m;
    _Float16* d = dataOut + base * 6;
    d[0] = (_Float16)(w * m);
    d[1] = (_Float16)((x - mx) * m);
    d[2] = (_Float16)((y - my) * m);
    d[3] = (_Float16)((z - mz) * m);
    d[4] = (_Float16)((x - ccx) * m);
    d[5] = (_Float16)((y - ccy) * m);
  }
}

// ---------------------------------------------------------------------------
// K2: FPS (10 centroids / pillar); writes new_xyz f32 + f16 into SA1 ch 0..2
// ---------------------------------------------------------------------------
__global__ __launch_bounds__(128) void k_fps(
    const float* __restrict__ xyz, float* __restrict__ nxyz,
    _Float16* __restrict__ sa1In)
{
  const int p = blockIdx.x;
  const int t = threadIdx.x;
  __shared__ float xs[N_PTS], ys[N_PTS], zs[N_PTS], dist[N_PTS];
  __shared__ float rv[128];
  __shared__ int   ri[128];
  __shared__ int   sfar;
  if (t < N_PTS) {
    size_t b = (size_t)p * N_PTS + t;
    xs[t] = xyz[0 * PN + b]; ys[t] = xyz[1 * PN + b]; zs[t] = xyz[2 * PN + b];
    dist[t] = 1e10f;
  }
  if (t == 0) sfar = 0;
  __syncthreads();
  for (int i = 0; i < NCENT; ++i) {
    const int far = sfar;
    const float cx = xs[far], cy = ys[far], cz = zs[far];
    if (t == 0) {
      size_t ob = (size_t)p * NCENT + i;
      nxyz[ob * 3 + 0] = cx; nxyz[ob * 3 + 1] = cy; nxyz[ob * 3 + 2] = cz;
      _Float16* sp = sa1In + ob * SA1_CH;
      sp[0] = (_Float16)cx; sp[1] = (_Float16)cy; sp[2] = (_Float16)cz;
    }
    float v = -1e30f;
    if (t < N_PTS) {
      float dx = xs[t] - cx, dy = ys[t] - cy, dz = zs[t] - cz;
      float d = fminf(dist[t], dx * dx + dy * dy + dz * dz);
      dist[t] = d;
      v = d;
    }
    rv[t] = v; ri[t] = t;
    __syncthreads();
    for (int s = 64; s > 0; s >>= 1) {
      if (t < s) {
        float v2 = rv[t + s]; int i2 = ri[t + s];
        if (v2 > rv[t] || (v2 == rv[t] && i2 < ri[t])) { rv[t] = v2; ri[t] = i2; }
      }
      __syncthreads();
    }
    if (t == 0) sfar = ri[0];
    __syncthreads();
  }
}

// ---------------------------------------------------------------------------
// K3: fused ball-query + group + 4-layer WMMA MLP + k-max, per pillar
// ---------------------------------------------------------------------------
struct BranchCfg {
  int k, rows, Mpad, Mt;
  int nOut[4], kc[4], woff[4], boff[4];
  int chOff;
};

__global__ __launch_bounds__(256) void k_branch(
    const float* __restrict__ xyz, const _Float16* __restrict__ data,
    const float* __restrict__ nxyz,
    const _Float16* __restrict__ wAll, const float* __restrict__ bAll,
    _Float16* __restrict__ sa1In, BranchCfg cfg, float r2)
{
  extern __shared__ char smem[];
  _Float16* X  = (_Float16*)smem;                      // Mpad x 32
  _Float16* Aa = X + (size_t)cfg.Mpad * 32;            // Mpad x 128
  _Float16* Bb = Aa + (size_t)cfg.Mpad * 128;          // Mpad x 128
  float* lx = (float*)(Bb + (size_t)cfg.Mpad * 128);   // 100 (16B aligned)
  float* ly = lx + N_PTS;
  float* lz = ly + N_PTS;
  _Float16* ld = (_Float16*)(lz + N_PTS);              // 100*6 (16B aligned)
  float* lnx = (float*)(ld + N_PTS * 6);               // 10*3
  int* gidx = (int*)(lnx + NCENT * 3);                 // 10*k (<=100)

  const int p = blockIdx.x;
  const int t = threadIdx.x;

  // async DMA of the pillar's x/y/z planes (3 x 400B) + data (1200B) into LDS
  if (t < 150) {
    unsigned laddr; const char* gaddr;
    if (t < 25) {
      laddr = lds_addr(lx) + (unsigned)t * 16u;
      gaddr = (const char*)(xyz + 0 * PN + (size_t)p * N_PTS) + (size_t)t * 16;
    } else if (t < 50) {
      laddr = lds_addr(ly) + (unsigned)(t - 25) * 16u;
      gaddr = (const char*)(xyz + 1 * PN + (size_t)p * N_PTS) + (size_t)(t - 25) * 16;
    } else if (t < 75) {
      laddr = lds_addr(lz) + (unsigned)(t - 50) * 16u;
      gaddr = (const char*)(xyz + 2 * PN + (size_t)p * N_PTS) + (size_t)(t - 50) * 16;
    } else {
      laddr = lds_addr(ld) + (unsigned)(t - 75) * 16u;
      gaddr = (const char*)(data + (size_t)p * N_PTS * 6) + (size_t)(t - 75) * 16;
    }
    async_b128(laddr, gaddr);
  }
  // overlap: centroid load + zero activation buffers while DMA is in flight
  for (int i = t; i < NCENT * 3; i += 256) lnx[i] = nxyz[(size_t)p * NCENT * 3 + i];
  for (int i = t; i < cfg.Mpad * 128; i += 256) { Aa[i] = (_Float16)0.f; Bb[i] = (_Float16)0.f; }
  async_wait0();
  __syncthreads();

  // ball query: first k points (ascending index) within radius
  if (t < NCENT) {
    const float qx = lnx[t * 3], qy = lnx[t * 3 + 1], qz = lnx[t * 3 + 2];
    int cnt = 0;
    for (int j = 0; j < N_PTS && cnt < cfg.k; ++j) {
      float dx = lx[j] - qx, dy = ly[j] - qy, dz = lz[j] - qz;
      if (dx * dx + dy * dy + dz * dz <= r2) gidx[t * cfg.k + cnt++] = j;
    }
    int first = (cnt > 0) ? gidx[t * cfg.k] : (N_PTS - 1);
    for (int j = cnt; j < cfg.k; ++j) gidx[t * cfg.k + j] = first;
  }
  __syncthreads();

  // build grouped activation matrix X (rows x 9, padded to Mpad x 32)
  for (int row = t; row < cfg.Mpad; row += 256) {
    _Float16* xr = X + (size_t)row * 32;
    if (row < cfg.rows) {
      int s = row / cfg.k, j = row - s * cfg.k;
      int id = gidx[s * cfg.k + j];
#pragma unroll
      for (int c = 0; c < 6; ++c) xr[c] = ld[id * 6 + c];
      xr[6] = (_Float16)(lx[id] - lnx[s * 3 + 0]);
      xr[7] = (_Float16)(ly[id] - lnx[s * 3 + 1]);
      xr[8] = (_Float16)(lz[id] - lnx[s * 3 + 2]);
#pragma unroll
      for (int c = 9; c < 32; ++c) xr[c] = (_Float16)0.f;
    } else {
#pragma unroll
      for (int c = 0; c < 32; ++c) xr[c] = (_Float16)0.f;
    }
  }
  __syncthreads();

  const _Float16* in = X;
  int inStride = 32;
  _Float16* out = Aa;
  for (int l = 0; l < 4; ++l) {
    block_gemm_relu(in, inStride, wAll + cfg.woff[l], cfg.kc[l] * 32,
                    bAll + cfg.boff[l], out, 128,
                    cfg.Mt, cfg.nOut[l] / 16, cfg.kc[l]);
    __syncthreads();
    in = out; inStride = 128;
    out = (out == Aa) ? Bb : Aa;
  }

  // max over k samples per centroid, write f16 into SA1 input channels
  const int nLast = cfg.nOut[3];
  for (int i = t; i < NCENT * nLast; i += 256) {
    int s = i / nLast, c = i - s * nLast;
    float mv = -1e30f;
    for (int j = 0; j < cfg.k; ++j)
      mv = fmaxf(mv, (float)in[(size_t)(s * cfg.k + j) * 128 + c]);
    sa1In[((size_t)p * NCENT + s) * SA1_CH + cfg.chOff + c] = (_Float16)mv;
  }
}

// ---------------------------------------------------------------------------
// K4: SA1 fused 3-layer WMMA MLP (16 pillars -> 160 rows) + max over 10 rows
// ---------------------------------------------------------------------------
struct Sa1Cfg { int woff[3]; int boff[3]; };

__global__ __launch_bounds__(256) void k_sa1(
    const _Float16* __restrict__ sa1In,
    const _Float16* __restrict__ wAll, const float* __restrict__ bAll,
    float* __restrict__ out, Sa1Cfg cfg)
{
  extern __shared__ char smem[];
  _Float16* Aa = (_Float16*)smem;         // 160 x 288
  _Float16* Bb = Aa + 160 * SA1_CH;       // 160 x 512
  const int p0 = blockIdx.x * 16;
  const int t = threadIdx.x;

  // async DMA of the 160x288 f16 activation tile (92160B = 5760 B128 chunks)
  {
    const unsigned ldsA = lds_addr(Aa);
    const char* g = (const char*)(sa1In + (size_t)p0 * NCENT * SA1_CH);
    for (int i = t; i < (160 * SA1_CH * 2) / 16; i += 256)
      async_b128(ldsA + (unsigned)i * 16u, g + (size_t)i * 16);
    // warm L2/WGP$ with the first weight tile while the DMA is in flight
    __builtin_prefetch(wAll + cfg.woff[0] + ((size_t)t << 6), 0, 0);
    async_wait0();
  }
  __syncthreads();

  // L1: 288 -> 512
  block_gemm_relu(Aa, SA1_CH, wAll + cfg.woff[0], 288, bAll + cfg.boff[0],
                  Bb, 512, 10, 32, 9);
  __syncthreads();
  // L2: 512 -> 256 (into Aa, stride 288; only cols 0..255 read next)
  block_gemm_relu(Bb, 512, wAll + cfg.woff[1], 512, bAll + cfg.boff[1],
                  Aa, SA1_CH, 10, 16, 16);
  __syncthreads();
  // L3: 256 -> 128 (into Bb, stride 512)
  block_gemm_relu(Aa, SA1_CH, wAll + cfg.woff[2], 256, bAll + cfg.boff[2],
                  Bb, 512, 10, 8, 8);
  __syncthreads();

  // max over the 10 centroids per pillar -> f32 output
  for (int i = t; i < 16 * 128; i += 256) {
    int pi = i >> 7, c = i & 127;
    float mv = -1e30f;
    for (int r = 0; r < NCENT; ++r)
      mv = fmaxf(mv, (float)Bb[(size_t)(pi * NCENT + r) * 512 + c]);
    out[((size_t)(p0 + pi)) * 128 + c] = mv;
  }
}

// ---------------------------------------------------------------------------
// Host: layer table, ws carving, launches
// ---------------------------------------------------------------------------
struct LayerDefH { int N, K; };
static const LayerDefH g_layers[15] = {
  {16, 9}, {32, 16}, {48, 32}, {64, 48},     // sa0 branch 0
  {32, 9}, {32, 32}, {48, 32}, {64, 48},     // sa0 branch 1
  {32, 9}, {64, 32}, {96, 64}, {128, 96},    // sa0 branch 2
  {512, 259}, {256, 512}, {128, 256}         // sa1
};

extern "C" void kernel_launch(void* const* d_in, const int* in_sizes, int n_in,
                              void* d_out, int out_size, void* d_ws, size_t ws_size,
                              hipStream_t stream) {
  (void)out_size; (void)ws_size;
  // -- locate inputs by size (robust to param pytree flatten order) --
  int prm[90]; int np = 0;
  int featI = -1, nvI = -1, coI = -1;
  for (int i = 0; i < n_in; ++i) {
    int s = in_sizes[i];
    if (s == P_TOT * N_PTS * C_IN_ && featI < 0) { featI = i; continue; }
    if (s == P_TOT && nvI < 0) { nvI = i; continue; }
    if (s == P_TOT * 4 && coI < 0) { coI = i; continue; }
    if (np < 90) prm[np++] = i;
  }
  const float* feat = (const float*)d_in[featI < 0 ? 0 : featI];
  const int* nvox   = (const int*)d_in[nvI < 0 ? 0 : nvI];
  const int* coors  = (const int*)d_in[coI < 0 ? 0 : coI];

  // -- workspace carve --
  char* ws = (char*)d_ws;
  size_t off = 0;
  auto carve = [&](size_t bytes) -> char* {
    char* r = ws + off;
    off = (off + bytes + 255) & ~(size_t)255;
    return r;
  };
  float*    xyzBuf  = (float*)carve(PN * 3 * 4);               // 3 SoA planes
  _Float16* dataBuf = (_Float16*)carve(PN * 6 * 2);
  float*    nxyzBuf = (float*)carve((size_t)P_TOT * NCENT * 3 * 4);
  _Float16* sa1In   = (_Float16*)carve((size_t)P_TOT * NCENT * SA1_CH * 2);

  int woff[15], boff[15];
  int wacc = 0, bacc = 0;
  for (int l = 0; l < 15; ++l) {
    woff[l] = wacc; boff[l] = bacc;
    int kpad = ((g_layers[l].K + 31) / 32) * 32;
    wacc += g_layers[l].N * kpad;
    bacc += g_layers[l].N;
  }
  _Float16* wFold = (_Float16*)carve((size_t)wacc * 2);
  float*    bFold = (float*)carve((size_t)bacc * 4);

  // -- fold BN into weights/bias --
  for (int l = 0; l < 15; ++l) {
    const int N = g_layers[l].N, K = g_layers[l].K;
    const int kpad = ((K + 31) / 32) * 32;
    const float* pw = nullptr;
    const float* po[5] = {nullptr, nullptr, nullptr, nullptr, nullptr};
    int no = 0;
    for (int j = 0; j < 6; ++j) {
      int gi = prm[l * 6 + j];
      if (in_sizes[gi] == N * K && !pw) pw = (const float*)d_in[gi];
      else if (no < 5) po[no++] = (const float*)d_in[gi];
    }
    if (!pw) pw = (const float*)d_in[prm[l * 6]];
    for (int j = no; j < 5; ++j) po[j] = pw;
    const int tot = N * kpad;
    k_fold<<<(tot + 255) / 256, 256, 0, stream>>>(
        pw, po[0], po[1], po[2], po[3], po[4], N, K, kpad,
        wFold + woff[l], bFold + boff[l]);
  }

  // -- clear SA1 input (covers padded channels 259..287) --
  k_zero<<<4096, 256, 0, stream>>>(sa1In, (size_t)P_TOT * NCENT * SA1_CH / 8);

  // -- preprocess + FPS --
  k_pre<<<P_TOT, 128, 0, stream>>>(feat, nvox, coors, xyzBuf, dataBuf);
  k_fps<<<P_TOT, 128, 0, stream>>>(xyzBuf, nxyzBuf, sa1In);

  // -- MSG branches --
  const float radii[3] = {0.32f, 0.36f, 0.40f};
  const int   ks[3]    = {6, 8, 10};
  const int   chO[3]   = {3, 3 + 64, 3 + 128};
  for (int br = 0; br < 3; ++br) {
    BranchCfg c{};
    c.k = ks[br];
    c.rows = NCENT * ks[br];
    c.Mpad = ((c.rows + 15) / 16) * 16;
    c.Mt = c.Mpad / 16;
    for (int l = 0; l < 4; ++l) {
      int gl = br * 4 + l;
      c.nOut[l] = g_layers[gl].N;
      c.kc[l]   = (g_layers[gl].K + 31) / 32;
      c.woff[l] = woff[gl];
      c.boff[l] = boff[gl];
    }
    c.chOff = chO[br];
    size_t sm = (size_t)(c.Mpad * 32 + 2 * c.Mpad * 128) * 2   // X + 2 act bufs
              + N_PTS * 3 * 4                                   // lx/ly/lz
              + N_PTS * 6 * 2                                   // data
              + NCENT * 3 * 4                                   // new_xyz
              + NCENT * 10 * 4 + 64;                            // gidx + pad
    k_branch<<<P_TOT, 256, sm, stream>>>(xyzBuf, dataBuf, nxyzBuf,
                                         wFold, bFold, sa1In, c,
                                         radii[br] * radii[br]);
  }

  // -- SA1 --
  Sa1Cfg s{};
  for (int l = 0; l < 3; ++l) { s.woff[l] = woff[12 + l]; s.boff[l] = boff[12 + l]; }
  size_t smSa1 = (size_t)(160 * SA1_CH + 160 * 512) * 2;  // 256 KB LDS (<=320KB WGP)
  k_sa1<<<P_TOT / 16, 256, smSa1, stream>>>(sa1In, wFold, bFold, (float*)d_out, s);
}
